// CustomFlashAttentionPyCore_49452253446942
// MI455X (gfx1250) — compile-verified, matrix operand order checked by reading, not run on hardware
//
#include <hip/hip_runtime.h>

// Flash attention fwd, causal, B=1 H=16 S=2048 D=64, fp32 in/out.
// Core math on v_wmma_f32_16x16x32_f16 (f16 operands, f32 accumulate).
// Prepass converts K -> f16 (same layout) and V -> f16 transposed [h][d][s]
// in workspace so the hot loop does pure b128 f16 loads + WMMA.

typedef _Float16 v16h  __attribute__((ext_vector_type(16)));
typedef _Float16 h8    __attribute__((ext_vector_type(8)));
typedef _Float16 h4    __attribute__((ext_vector_type(4)));
typedef float    v8f   __attribute__((ext_vector_type(8)));
typedef float    f32x4 __attribute__((ext_vector_type(4)));

#define SEQ 2048
#define NH  16
#define DH  64
#define ELEMS ((size_t)NH * SEQ * DH)

// swap data between lane halves (lane ^ 16): ds_swizzle SWAPX16 (xor=0x10,and=0x1f)
__device__ __forceinline__ float swap_half(float x) {
  return __int_as_float(__builtin_amdgcn_ds_swizzle(__float_as_int(x), 0x401F));
}
// broadcast from lane `idx` (0..15; both halves hold identical per-query stats)
__device__ __forceinline__ float bcast_row(float x, int idx) {
  return __int_as_float(__builtin_amdgcn_ds_bpermute(idx << 2, __float_as_int(x)));
}

// fp32 row -> f16 operand in the wave32 16-bit A/B per-lane pattern:
// elements 0..7 <- rowp[off..off+7], 8..15 <- rowp[off+16..off+23]
__device__ __forceinline__ v16h load_row_pattern_f32(const float* __restrict__ rowp,
                                                     int off) {
  f32x4 a0 = *(const f32x4*)(rowp + off + 0);
  f32x4 a1 = *(const f32x4*)(rowp + off + 4);
  f32x4 a2 = *(const f32x4*)(rowp + off + 16);
  f32x4 a3 = *(const f32x4*)(rowp + off + 20);
  v16h r;
#pragma unroll
  for (int i = 0; i < 4; ++i) {
    r[i]      = (_Float16)a0[i];
    r[4 + i]  = (_Float16)a1[i];
    r[8 + i]  = (_Float16)a2[i];
    r[12 + i] = (_Float16)a3[i];
  }
  return r;
}

// f16 row -> operand pattern: two contiguous 16B loads, no conversion
__device__ __forceinline__ v16h load_row_pattern_f16(const _Float16* __restrict__ rowp,
                                                     int off) {
  h8 a = *(const h8*)(rowp + off);
  h8 b = *(const h8*)(rowp + off + 16);
  v16h r;
#pragma unroll
  for (int i = 0; i < 8; ++i) {
    r[i]     = a[i];
    r[8 + i] = b[i];
  }
  return r;
}

__device__ __forceinline__ v8f wmma_f16(v16h a, v16h b, v8f c) {
  return __builtin_amdgcn_wmma_f32_16x16x32_f16(false, a, false, b, (short)0, c,
                                                false, false);
}

// ---------------- prepass kernels ----------------

// K (fp32, [h][s][d]) -> K16 (f16, same layout), elementwise
__global__ __launch_bounds__(256)
void cvt_k16(const float* __restrict__ in, _Float16* __restrict__ out) {
  const size_t i = ((size_t)blockIdx.x * blockDim.x + threadIdx.x) * 4;
  f32x4 v = *(const f32x4*)(in + i);
  h4 o;
#pragma unroll
  for (int j = 0; j < 4; ++j) o[j] = (_Float16)v[j];
  *(h4*)(out + i) = o;
}

// V (fp32, [h][s][d]) -> VT16 (f16, [h][d][s]) via LDS 32x32 tile transpose
__global__ __launch_bounds__(256)
void cvt_vt16(const float* __restrict__ V, _Float16* __restrict__ VT) {
  __shared__ _Float16 tile[32][33];
  const int h  = blockIdx.z;
  const int s0 = blockIdx.x * 32;
  const int d0 = blockIdx.y * 32;
  const int tx = threadIdx.x;   // 0..31
  const int ty = threadIdx.y;   // 0..7
  const float* __restrict__ Vh = V + (size_t)h * SEQ * DH;
  _Float16* __restrict__ VTh   = VT + (size_t)h * DH * SEQ;
#pragma unroll
  for (int k = 0; k < 4; ++k) {
    const int s = ty + 8 * k;
    tile[s][tx] = (_Float16)Vh[(size_t)(s0 + s) * DH + d0 + tx];
  }
  __syncthreads();
#pragma unroll
  for (int k = 0; k < 4; ++k) {
    const int d = ty + 8 * k;
    VTh[(size_t)(d0 + d) * SEQ + s0 + tx] = tile[tx][d];
  }
}

// ---------------- main attention kernel (f16-staged K/V) ----------------

__global__ __launch_bounds__(128)
void fa_fwd_causal_f16(const float* __restrict__ Q,
                       const _Float16* __restrict__ K16,
                       const _Float16* __restrict__ VT16,
                       float* __restrict__ O) {
  const int lane = threadIdx.x & 31;
  const int wave = threadIdx.x >> 5;
  const int lq   = lane & 15;   // query column within 16x16 tiles
  const int hi   = lane >> 4;   // lane half
  const int h    = blockIdx.y;
  const int q0   = blockIdx.x * 64 + wave * 16;

  const float* __restrict__ Qh     = Q + (size_t)h * SEQ * DH;
  const _Float16* __restrict__ Kh  = K16 + (size_t)h * SEQ * DH;
  const _Float16* __restrict__ VTh = VT16 + (size_t)h * DH * SEQ;
  float* __restrict__ Oh           = O + (size_t)h * SEQ * DH;

  // Q as B-operand of S^T = K * Q^T (per-lane data == A-pattern of row q0+lq)
  const float* qrow = Qh + (size_t)(q0 + lq) * DH;
  const v16h qb0 = load_row_pattern_f32(qrow, 8 * hi);
  const v16h qb1 = load_row_pattern_f32(qrow, 32 + 8 * hi);

  v8f oacc[4] = {v8f{}, v8f{}, v8f{}, v8f{}};
  float m_run = -1e37f;
  float l_run = 0.0f;

  const float sc = 0.125f * 1.44269504088896f;  // 1/sqrt(64) * log2(e)

  const int nchunks = (q0 + 16 + 31) >> 5;
  for (int ch = 0; ch < nchunks; ++ch) {
    const int kb = ch << 5;

    // ---- issue ALL of this chunk's loads up front (16x global_load_b128):
    // the S^T WMMAs then retire on partial load counts, and the V operands'
    // latency is hidden under the softmax VALU section.
    const _Float16* kr0 = Kh + (size_t)(kb + lq) * DH;
    const _Float16* kr1 = Kh + (size_t)(kb + 16 + lq) * DH;
    const v16h ka00 = load_row_pattern_f16(kr0, 8 * hi);
    const v16h ka01 = load_row_pattern_f16(kr0, 32 + 8 * hi);
    const v16h ka10 = load_row_pattern_f16(kr1, 8 * hi);
    const v16h ka11 = load_row_pattern_f16(kr1, 32 + 8 * hi);
    v16h vb[4];
#pragma unroll
    for (int nt = 0; nt < 4; ++nt) {
      vb[nt] = load_row_pattern_f16(VTh + (size_t)(nt * 16 + lq) * SEQ + kb, 8 * hi);
    }

    if (ch + 1 < nchunks) {  // pull next chunk toward the WGP
      __builtin_prefetch(Kh + (size_t)(kb + 32 + lq) * DH, 0, 3);
      __builtin_prefetch(VTh + (size_t)lq * SEQ + kb + 32, 0, 3);
    }

    // ---- S^T = K * Q^T : two 16-key tiles, d=64 split in two K=32 steps
    v8f st0 = {}, st1 = {};
    st0 = wmma_f16(ka00, qb0, st0);
    st0 = wmma_f16(ka01, qb1, st0);
    st1 = wmma_f16(ka10, qb0, st1);
    st1 = wmma_f16(ka11, qb1, st1);

    // ---- online softmax for query q = q0+lq; lane holds keys kb+16t+r+8*hi
    float s[16];
#pragma unroll
    for (int r = 0; r < 8; ++r) {
      s[r]     = st0[r] * sc;
      s[8 + r] = st1[r] * sc;
    }
    if (kb + 31 > q0) {  // chunk crosses the diagonal -> mask
      const int q = q0 + lq;
#pragma unroll
      for (int r = 0; r < 8; ++r) {
        if (kb + r + 8 * hi > q)      s[r]     = -1e37f;
        if (kb + 16 + r + 8 * hi > q) s[8 + r] = -1e37f;
      }
    }

    float mx = s[0];
#pragma unroll
    for (int i = 1; i < 16; ++i) mx = fmaxf(mx, s[i]);
    mx = fmaxf(mx, swap_half(mx));

    const float m_new = fmaxf(m_run, mx);
    const float corr  = exp2f(m_run - m_new);

    float p[16];
    float sum = 0.0f;
#pragma unroll
    for (int i = 0; i < 16; ++i) {
      p[i] = exp2f(s[i] - m_new);
      sum += p[i];
    }
    sum += swap_half(sum);
    l_run = fmaf(corr, l_run, sum);
    m_run = m_new;

#pragma unroll
    for (int r = 0; r < 8; ++r) {
      const float cb = bcast_row(corr, r + 8 * hi);
      oacc[0][r] *= cb;
      oacc[1][r] *= cb;
      oacc[2][r] *= cb;
      oacc[3][r] *= cb;
    }

    // P is already in the f16 A-operand layout per lane
    v16h pa;
#pragma unroll
    for (int i = 0; i < 16; ++i) pa[i] = (_Float16)p[i];

    // ---- O += P * V (V operands loaded at chunk start, long since arrived)
#pragma unroll
    for (int nt = 0; nt < 4; ++nt) {
      oacc[nt] = wmma_f16(pa, vb[nt], oacc[nt]);
    }
  }

  const float linv = 1.0f / l_run;
#pragma unroll
  for (int r = 0; r < 8; ++r) {
    const float lb = bcast_row(linv, r + 8 * hi);
    float* __restrict__ orow = Oh + (size_t)(q0 + r + 8 * hi) * DH;
    orow[0 + lq]  = oacc[0][r] * lb;
    orow[16 + lq] = oacc[1][r] * lb;
    orow[32 + lq] = oacc[2][r] * lb;
    orow[48 + lq] = oacc[3][r] * lb;
  }
}

// ---------------- fallback: direct-from-fp32 (no workspace) ----------------

__global__ __launch_bounds__(128)
void fa_fwd_causal_f32(const float* __restrict__ Q, const float* __restrict__ K,
                       const float* __restrict__ V, float* __restrict__ O) {
  const int lane = threadIdx.x & 31;
  const int wave = threadIdx.x >> 5;
  const int lq   = lane & 15;
  const int hi   = lane >> 4;
  const int h    = blockIdx.y;
  const int q0   = blockIdx.x * 64 + wave * 16;

  const float* __restrict__ Qh = Q + (size_t)h * SEQ * DH;
  const float* __restrict__ Kh = K + (size_t)h * SEQ * DH;
  const float* __restrict__ Vh = V + (size_t)h * SEQ * DH;
  float* __restrict__ Oh       = O + (size_t)h * SEQ * DH;

  const float* qrow = Qh + (size_t)(q0 + lq) * DH;
  const v16h qb0 = load_row_pattern_f32(qrow, 8 * hi);
  const v16h qb1 = load_row_pattern_f32(qrow, 32 + 8 * hi);

  v8f oacc[4] = {v8f{}, v8f{}, v8f{}, v8f{}};
  float m_run = -1e37f;
  float l_run = 0.0f;
  const float sc = 0.125f * 1.44269504088896f;

  const int nchunks = (q0 + 16 + 31) >> 5;
  for (int ch = 0; ch < nchunks; ++ch) {
    const int kb = ch << 5;
    v8f st0 = {}, st1 = {};
    {
      const float* kr = Kh + (size_t)(kb + lq) * DH;
      st0 = wmma_f16(load_row_pattern_f32(kr, 8 * hi), qb0, st0);
      st0 = wmma_f16(load_row_pattern_f32(kr, 32 + 8 * hi), qb1, st0);
    }
    {
      const float* kr = Kh + (size_t)(kb + 16 + lq) * DH;
      st1 = wmma_f16(load_row_pattern_f32(kr, 8 * hi), qb0, st1);
      st1 = wmma_f16(load_row_pattern_f32(kr, 32 + 8 * hi), qb1, st1);
    }
    float s[16];
#pragma unroll
    for (int r = 0; r < 8; ++r) {
      s[r]     = st0[r] * sc;
      s[8 + r] = st1[r] * sc;
    }
    if (kb + 31 > q0) {
      const int q = q0 + lq;
#pragma unroll
      for (int r = 0; r < 8; ++r) {
        if (kb + r + 8 * hi > q)      s[r]     = -1e37f;
        if (kb + 16 + r + 8 * hi > q) s[8 + r] = -1e37f;
      }
    }
    float mx = s[0];
#pragma unroll
    for (int i = 1; i < 16; ++i) mx = fmaxf(mx, s[i]);
    mx = fmaxf(mx, swap_half(mx));
    const float m_new = fmaxf(m_run, mx);
    const float corr  = exp2f(m_run - m_new);
    float p[16];
    float sum = 0.0f;
#pragma unroll
    for (int i = 0; i < 16; ++i) {
      p[i] = exp2f(s[i] - m_new);
      sum += p[i];
    }
    sum += swap_half(sum);
    l_run = fmaf(corr, l_run, sum);
    m_run = m_new;
#pragma unroll
    for (int r = 0; r < 8; ++r) {
      const float cb = bcast_row(corr, r + 8 * hi);
      oacc[0][r] *= cb;
      oacc[1][r] *= cb;
      oacc[2][r] *= cb;
      oacc[3][r] *= cb;
    }
    v16h pa;
#pragma unroll
    for (int i = 0; i < 16; ++i) pa[i] = (_Float16)p[i];
    const float* __restrict__ vb_base = Vh + (size_t)kb * DH;
#pragma unroll
    for (int nt = 0; nt < 4; ++nt) {
      const int col = nt * 16 + lq;
      v16h vb;
#pragma unroll
      for (int e = 0; e < 16; ++e) {
        const int kk = e + 8 * hi + (e & 8);
        vb[e] = (_Float16)vb_base[(size_t)kk * DH + col];
      }
      oacc[nt] = wmma_f16(pa, vb, oacc[nt]);
    }
  }
  const float linv = 1.0f / l_run;
#pragma unroll
  for (int r = 0; r < 8; ++r) {
    const float lb = bcast_row(linv, r + 8 * hi);
    float* __restrict__ orow = Oh + (size_t)(q0 + r + 8 * hi) * DH;
    orow[0 + lq]  = oacc[0][r] * lb;
    orow[16 + lq] = oacc[1][r] * lb;
    orow[32 + lq] = oacc[2][r] * lb;
    orow[48 + lq] = oacc[3][r] * lb;
  }
}

extern "C" void kernel_launch(void* const* d_in, const int* in_sizes, int n_in,
                              void* d_out, int out_size, void* d_ws, size_t ws_size,
                              hipStream_t stream) {
  (void)in_sizes; (void)n_in; (void)out_size;
  const float* q = (const float*)d_in[0];
  const float* k = (const float*)d_in[1];
  const float* v = (const float*)d_in[2];
  float* out = (float*)d_out;

  dim3 grid(SEQ / 64, NH);
  dim3 block(128);

  const size_t need = 2 * ELEMS * sizeof(_Float16);  // K16 + VT16 = 8 MB
  if (ws_size >= need) {
    _Float16* k16  = (_Float16*)d_ws;
    _Float16* vt16 = k16 + ELEMS;
    cvt_k16<<<(unsigned)(ELEMS / (256 * 4)), 256, 0, stream>>>(k, k16);
    cvt_vt16<<<dim3(SEQ / 32, DH / 32, NH), dim3(32, 8), 0, stream>>>(v, vt16);
    fa_fwd_causal_f16<<<grid, block, 0, stream>>>(q, k16, vt16, out);
  } else {
    fa_fwd_causal_f32<<<grid, block, 0, stream>>>(q, k, v, out);
  }
}
